// MultiHeadedAttention_15135464751220
// MI455X (gfx1250) — matrix operand order repaired and checked by reading, not compiled, not run on hardware
//
#include <hip/hip_runtime.h>
#include <hip/hip_bf16.h>

#define H_   12
#define S_   111
#define DKX  64
#define DX   768
#define BX   256
#define M_   (BX * S_)   // 28416 rows, divisible by 128

typedef __attribute__((ext_vector_type(16))) __bf16 v16bf;
typedef __attribute__((ext_vector_type(8)))  __bf16 v8bf;
typedef __attribute__((ext_vector_type(8)))  float  v8f;

// ---- WMMA fragment helpers (CDNA5 16x16x32 bf16 layouts) ----
// A 16x32: lane(0..15)=row M, half selects K-phase: elems 0..7 = K[8*half .. 8*half+7],
//          elems 8..15 = K[16+8*half .. 16+8*half+7]  (two 16B contiguous loads)
__device__ __forceinline__ v16bf load_a_frag(const __bf16* rowK, int half) {
  v8bf lo = *reinterpret_cast<const v8bf*>(rowK + 8 * half);
  v8bf hi = *reinterpret_cast<const v8bf*>(rowK + 16 + 8 * half);
  return __builtin_shufflevector(lo, hi, 0,1,2,3,4,5,6,7,8,9,10,11,12,13,14,15);
}
// B 32x16: lane(0..15)=col N, elems 0..7 = K[8*half .. 8*half+7],
//          elems 8..15 = K[16+8*half .. 16+8*half+7]  (two 16B contiguous loads)
__device__ __forceinline__ v16bf load_b_frag(const __bf16* colK, int half) {
  v8bf lo = *reinterpret_cast<const v8bf*>(colK + 8 * half);
  v8bf hi = *reinterpret_cast<const v8bf*>(colK + 16 + 8 * half);
  return __builtin_shufflevector(lo, hi, 0,1,2,3,4,5,6,7,8,9,10,11,12,13,14,15);
}
__device__ __forceinline__ v8f wmma_bf16(v16bf a, v16bf b, v8f c) {
  return __builtin_amdgcn_wmma_f32_16x16x32_bf16(false, a, false, b, (short)0, c, false, false);
}

// ---- GEMM: out = X(28416x768) @ W^T(768x768) + bias ----
// HEAD_OUT: scatter to [B, H, S, 64] (bf16 q/k/v), else row-major [M,768]
template <typename TIn, typename TOut, bool HEAD_OUT>
__global__ __launch_bounds__(256)
void gemm_xwt_kernel(const TIn* __restrict__ X, const float* __restrict__ W,
                     const float* __restrict__ bias, TOut* __restrict__ out) {
  constexpr int BM = 128, BN = 64, BK = 32, LDL = 48;   // 96B row stride: 16B & 32B aligned
  __shared__ __align__(32) __bf16 Xs[BM][LDL];
  __shared__ __align__(32) __bf16 Ws[BN][LDL];
  const int tid  = threadIdx.x;
  const int wave = tid >> 5, lane = tid & 31;
  const int l = lane & 15, half = lane >> 4;
  const int m0 = blockIdx.x * BM;
  const int n0 = blockIdx.y * BN;
  const int wm = (wave & 3) * 32, wn = (wave >> 2) * 32;  // 4x2 wave grid, 32x32 per wave
  v8f acc[2][2] = {};
  for (int k0 = 0; k0 < DX; k0 += BK) {
    #pragma unroll
    for (int i = tid; i < BM * BK; i += 256) {
      int r = i >> 5, c = i & 31;
      Xs[r][c] = (__bf16)(float)X[(size_t)(m0 + r) * DX + k0 + c];
    }
    #pragma unroll
    for (int i = tid; i < BN * BK; i += 256) {
      int r = i >> 5, c = i & 31;
      Ws[r][c] = (__bf16)(float)W[(size_t)(n0 + r) * DX + k0 + c];
    }
    if (k0 + BK < DX)  // pull next X K-tile toward L2 (global_prefetch_b8)
      __builtin_prefetch(&X[(size_t)(m0 + (tid >> 1)) * DX + k0 + BK + (tid & 1) * 16], 0, 0);
    __syncthreads();
    #pragma unroll
    for (int sm = 0; sm < 2; ++sm) {
      v16bf a = load_a_frag(&Xs[wm + sm * 16 + l][0], half);
      #pragma unroll
      for (int sn = 0; sn < 2; ++sn) {
        v16bf b = load_b_frag(&Ws[wn + sn * 16 + l][0], half);
        acc[sm][sn] = wmma_bf16(a, b, acc[sm][sn]);
      }
    }
    __syncthreads();
  }
  #pragma unroll
  for (int sm = 0; sm < 2; ++sm) {
    #pragma unroll
    for (int sn = 0; sn < 2; ++sn) {
      const int n = n0 + wn + sn * 16 + l;
      const float bv = bias[n];
      #pragma unroll
      for (int i = 0; i < 8; ++i) {
        const int m = m0 + wm + sm * 16 + i + 8 * half;   // C layout: VGPR i -> row i (+8 for hi half)
        const float v = acc[sm][sn][i] + bv;
        if (HEAD_OUT) {
          const int bb = m / S_, s = m - bb * S_;
          const int hh = n >> 6, dk = n & 63;
          out[((((size_t)bb * H_ + hh) * S_ + s) << 6) + dk] = (TOut)v;
        } else {
          out[(size_t)m * DX + n] = (TOut)v;
        }
      }
    }
  }
}

// ---- factor[b,q,k] = 1 + softmax_over_q(where(mask1==0,-1e9,mask1)) * mask1 ----
__global__ __launch_bounds__(128)
void factor_kernel(const float* __restrict__ mask1, float* __restrict__ factor) {
  const int b = blockIdx.x;
  const int k = threadIdx.x;
  if (k >= S_) return;
  const float* m1 = mask1 + (size_t)b * S_ * S_;
  float mx = -3.0e38f;
  for (int q = 0; q < S_; ++q) {
    float v0 = m1[q * S_ + k];
    float v = (v0 == 0.f) ? -1e9f : v0;
    mx = fmaxf(mx, v);
  }
  float sum = 0.f;
  for (int q = 0; q < S_; ++q) {
    float v0 = m1[q * S_ + k];
    float v = (v0 == 0.f) ? -1e9f : v0;
    sum += __expf(v - mx);
  }
  const float inv = 1.0f / sum;
  for (int q = 0; q < S_; ++q) {
    float v0 = m1[q * S_ + k];
    float v = (v0 == 0.f) ? -1e9f : v0;
    factor[((size_t)b * S_ + q) * S_ + k] = 1.0f + __expf(v - mx) * inv * v0;
  }
}

// ---- attention: one block per (b,h); 7 waves, each owns a 16-row q tile ----
__global__ __launch_bounds__(224)
void attention_kernel(const __bf16* __restrict__ qw, const __bf16* __restrict__ kw,
                      const __bf16* __restrict__ vw, const float* __restrict__ factor,
                      const float* __restrict__ weight, const int* __restrict__ mask,
                      float* __restrict__ scores_out, __bf16* __restrict__ xw) {
  constexpr int SP = 112;   // keys padded to 7*16
  constexpr int KP = 128;   // P K-dim padded to 4*32
  __shared__ __align__(32) __bf16 ks[SP][DKX];       // [key][d]
  __shared__ __align__(32) __bf16 vsT[DKX][KP];      // [d][key] (transposed for B frags)
  __shared__ __align__(32) __bf16 ps[7][16][KP];     // per-wave P tile, bf16
  const int tid  = threadIdx.x;
  const int wave = tid >> 5, lane = tid & 31;
  const int l = lane & 15, half = lane >> 4;
  const int bh = blockIdx.x;
  const int b = bh / H_, h = bh - b * H_;
  const __bf16* qb = qw + (size_t)bh * S_ * DKX;
  const __bf16* kb = kw + (size_t)bh * S_ * DKX;
  const __bf16* vb = vw + (size_t)bh * S_ * DKX;

  for (int i = tid; i < SP * DKX; i += 224) {
    int s = i >> 6, d = i & 63;
    ks[s][d] = (s < S_) ? kb[s * DKX + d] : (__bf16)0.f;
  }
  for (int i = tid; i < DKX * KP; i += 224) {
    int d = i >> 7, s = i & 127;
    vsT[d][s] = (s < S_) ? vb[s * DKX + d] : (__bf16)0.f;
  }
  __syncthreads();

  // Q A-fragments straight from global (rows clamped; invalid rows never stored)
  const int q0 = wave * 16;
  int qr = q0 + l; if (qr > S_ - 1) qr = S_ - 1;
  const __bf16* qrow = qb + (size_t)qr * DKX;
  const v16bf aq0 = load_a_frag(qrow, half);
  const v16bf aq1 = load_a_frag(qrow + 32, half);

  // scores tiles: S = Q @ K^T  (B[d][n=key] column = ks row)
  v8f sc[7];
  #pragma unroll
  for (int t = 0; t < 7; ++t) {
    v16bf b0 = load_b_frag(&ks[t * 16 + l][0], half);
    v16bf b1 = load_b_frag(&ks[t * 16 + l][32], half);
    v8f c = {};
    c = wmma_bf16(aq0, b0, c);
    c = wmma_bf16(aq1, b1, c);
    sc[t] = c;
  }

  // scale * factor * weight, key-mask, emit pre-softmax scores
  const float scale = 0.125f;        // 1/sqrt(64)
  const float BIG_NEG = -3.0e38f;    // padding sentinel (finite: avoids NaN in softmax)
  #pragma unroll
  for (int t = 0; t < 7; ++t) {
    const int kc = t * 16 + l;
    const int mk = (kc < S_) ? mask[(size_t)b * S_ + kc] : 0;
    #pragma unroll
    for (int i = 0; i < 8; ++i) {
      const int qi = q0 + i + 8 * half;
      float v;
      if (qi < S_ && kc < S_) {
        const float f  = factor[((size_t)b * S_ + qi) * S_ + kc];
        const float wg = weight[qi * S_ + kc];
        v = (mk != 0) ? sc[t][i] * scale * f * wg : -1e9f;
        scores_out[((size_t)bh * S_ + qi) * S_ + kc] = v;
      } else {
        v = BIG_NEG;
      }
      sc[t][i] = v;
    }
  }

  // row softmax: columns of row (i,half) live on the 16 lanes of this half
  float inv[8];
  #pragma unroll
  for (int i = 0; i < 8; ++i) {
    float m = BIG_NEG;
    #pragma unroll
    for (int t = 0; t < 7; ++t) m = fmaxf(m, sc[t][i]);
    #pragma unroll
    for (int off = 1; off < 16; off <<= 1) m = fmaxf(m, __shfl_xor(m, off, 32));
    float s = 0.f;
    #pragma unroll
    for (int t = 0; t < 7; ++t) { float e = __expf(sc[t][i] - m); sc[t][i] = e; s += e; }
    #pragma unroll
    for (int off = 1; off < 16; off <<= 1) s += __shfl_xor(s, off, 32);
    inv[i] = 1.0f / s;
  }

  // P -> LDS (bf16), zero K padding [112,128)
  #pragma unroll
  for (int t = 0; t < 7; ++t) {
    const int kc = t * 16 + l;
    #pragma unroll
    for (int i = 0; i < 8; ++i)
      ps[wave][i + 8 * half][kc] = (__bf16)(sc[t][i] * inv[i]);
  }
  #pragma unroll
  for (int i = 0; i < 8; ++i)
    ps[wave][i + 8 * half][112 + l] = (__bf16)0.f;
  __syncthreads();

  // X = P(16x112) @ V(112x64): 4 K-chunks x 4 N-tiles
  v8f xacc[4] = {};
  #pragma unroll
  for (int c = 0; c < 4; ++c) {
    v16bf ap = load_a_frag(&ps[wave][l][c * 32], half);
    #pragma unroll
    for (int dt = 0; dt < 4; ++dt) {
      v16bf bv2 = load_b_frag(&vsT[dt * 16 + l][c * 32], half);
      xacc[dt] = wmma_bf16(ap, bv2, xacc[dt]);
    }
  }
  #pragma unroll
  for (int dt = 0; dt < 4; ++dt) {
    const int d = dt * 16 + l;
    #pragma unroll
    for (int i = 0; i < 8; ++i) {
      const int qi = q0 + i + 8 * half;
      if (qi < S_)
        xw[(size_t)(b * S_ + qi) * DX + h * DKX + d] = (__bf16)xacc[dt][i];
    }
  }
}

extern "C" void kernel_launch(void* const* d_in, const int* in_sizes, int n_in,
                              void* d_out, int out_size, void* d_ws, size_t ws_size,
                              hipStream_t stream) {
  (void)in_sizes; (void)n_in; (void)out_size; (void)ws_size;
  const float* query  = (const float*)d_in[0];
  const float* key    = (const float*)d_in[1];
  const float* value  = (const float*)d_in[2];
  const int*   mask   = (const int*)  d_in[3];
  const float* mask1  = (const float*)d_in[4];
  const float* Wq     = (const float*)d_in[5];
  const float* bq     = (const float*)d_in[6];
  const float* Wk     = (const float*)d_in[7];
  const float* bk     = (const float*)d_in[8];
  const float* Wv     = (const float*)d_in[9];
  const float* bv     = (const float*)d_in[10];
  const float* Wo     = (const float*)d_in[11];
  const float* bo     = (const float*)d_in[12];
  const float* weight = (const float*)d_in[13];

  auto al = [](size_t x) { return (x + 255) & ~(size_t)255; };
  char* w = (char*)d_ws;
  const size_t nElem = (size_t)M_ * DX;               // 28416*768
  __bf16* qw = (__bf16*)w; w += al(nElem * sizeof(__bf16));
  __bf16* kw = (__bf16*)w; w += al(nElem * sizeof(__bf16));
  __bf16* vw = (__bf16*)w; w += al(nElem * sizeof(__bf16));
  __bf16* xw = (__bf16*)w; w += al(nElem * sizeof(__bf16));
  float* factor = (float*)w;                          // B*S*S floats

  float* ret    = (float*)d_out;
  float* scores = (float*)d_out + nElem;              // [B,H,S,S]

  dim3 gg(M_ / 128, DX / 64), gb(256);
  gemm_xwt_kernel<float, __bf16, true><<<gg, gb, 0, stream>>>(query, Wq, bq, qw);
  gemm_xwt_kernel<float, __bf16, true><<<gg, gb, 0, stream>>>(key,   Wk, bk, kw);
  gemm_xwt_kernel<float, __bf16, true><<<gg, gb, 0, stream>>>(value, Wv, bv, vw);
  factor_kernel<<<BX, 128, 0, stream>>>(mask1, factor);
  attention_kernel<<<BX * H_, 224, 0, stream>>>(qw, kw, vw, factor, weight, mask, scores, xw);
  gemm_xwt_kernel<__bf16, float, false><<<gg, gb, 0, stream>>>(xw, Wo, bo, ret);
}
